// SkelPointNet_19146964205685
// MI455X (gfx1250) — matrix operand order repaired and checked by reading, not compiled, not run on hardware
//
#include <hip/hip_runtime.h>
#include <hip/hip_bf16.h>

// ---------------------------------------------------------------------------
// Types for WMMA fragments (CDNA5 / gfx1250, wave32)
// ---------------------------------------------------------------------------
typedef _Float16 h8f __attribute__((ext_vector_type(8)));
typedef __attribute__((ext_vector_type(16))) _Float16 v16h;
typedef __attribute__((ext_vector_type(8)))  float    v8f;

#define BN_EPS 1e-5f

static inline int round32(int x) { return (x + 31) & ~31; }

// ---------------------------------------------------------------------------
// Farthest point sampling: one block per batch.
// xyz + running min-distance staged in LDS (32 KB of the 320 KB per WGP);
// per-iteration argmax via wave32 __shfl_xor (barrier-free) + one
// __syncthreads and an 8-way scalar combine. This is the latency-critical
// serial kernel, so minimizing barriers per iteration matters most.
// ---------------------------------------------------------------------------
#define FPS_THREADS 256
__global__ void fps_kernel(const float* __restrict__ xyz, int N, int npoint,
                           int* __restrict__ fidx)
{
    int b = blockIdx.x;
    const float* p = xyz + (size_t)b * N * 3;
    __shared__ float sx[2048], sy[2048], sz[2048];
    __shared__ float dist[2048];
    __shared__ float wrv[FPS_THREADS / 32];
    __shared__ int   wri[FPS_THREADS / 32];
    __shared__ int   s_last;
    int t = threadIdx.x;
    int lane = t & 31, wv = t >> 5;
    for (int j = t; j < N; j += FPS_THREADS) {
        sx[j] = p[j * 3 + 0];
        sy[j] = p[j * 3 + 1];
        sz[j] = p[j * 3 + 2];
        dist[j] = 1e10f;
    }
    if (t == 0) { s_last = 0; fidx[(size_t)b * npoint] = 0; }
    __syncthreads();
    for (int i = 1; i < npoint; ++i) {
        int last = s_last;
        float lx = sx[last], ly = sy[last], lz = sz[last];
        float best = -1.0f; int bi = 0x7fffffff;
        for (int j = t; j < N; j += FPS_THREADS) {
            float dx = sx[j] - lx;
            float dy = sy[j] - ly;
            float dz = sz[j] - lz;
            float d  = dx * dx + dy * dy + dz * dz;
            float dm = fminf(dist[j], d);
            dist[j] = dm;
            if (dm > best || (dm == best && j < bi)) { best = dm; bi = j; }
        }
        // wave32 argmax reduction, no barriers
        #pragma unroll
        for (int o = 16; o > 0; o >>= 1) {
            float ob = __shfl_xor(best, o, 32);
            int   oi = __shfl_xor(bi,   o, 32);
            if (ob > best || (ob == best && oi < bi)) { best = ob; bi = oi; }
        }
        if (lane == 0) { wrv[wv] = best; wri[wv] = bi; }
        __syncthreads();
        if (t == 0) {
            float bb = wrv[0]; int ii = wri[0];
            #pragma unroll
            for (int w = 1; w < FPS_THREADS / 32; ++w)
                if (wrv[w] > bb || (wrv[w] == bb && wri[w] < ii)) { bb = wrv[w]; ii = wri[w]; }
            s_last = ii; fidx[(size_t)b * npoint + i] = ii;
        }
        __syncthreads();
    }
}

// ---------------------------------------------------------------------------
// Gather sampled centers: new_xyz[b][s] = xyz[b][fidx[b][s]]
// ---------------------------------------------------------------------------
__global__ void gather_xyz_kernel(const float* __restrict__ xyz,
                                  const int* __restrict__ fidx,
                                  float* __restrict__ nxyz, int N, int np)
{
    int b = blockIdx.y;
    int s = blockIdx.x * blockDim.x + threadIdx.x;
    if (s >= np) return;
    int j = fidx[(size_t)b * np + s];
    #pragma unroll
    for (int d = 0; d < 3; ++d)
        nxyz[((size_t)b * np + s) * 3 + d] = xyz[((size_t)b * N + j) * 3 + d];
}

// ---------------------------------------------------------------------------
// Ball query (per batch): one WAVE per query point. 32 candidates per step;
// qualifying indices compacted in increasing order via ballot + prefix popc —
// exactly the reference's "first K indices within radius, padded with first".
// ---------------------------------------------------------------------------
__global__ void ball_query_kernel(const float* __restrict__ xyz,
                                  const float* __restrict__ nxyz,
                                  int N, int S, float r2, int K,
                                  int* __restrict__ idx)
{
    int s = blockIdx.x;
    if (s >= S) return;
    int lane = threadIdx.x;            // blockDim == 32 (one wave)
    float cx = nxyz[s * 3 + 0], cy = nxyz[s * 3 + 1], cz = nxyz[s * 3 + 2];
    int* row = idx + (size_t)s * K;
    int cnt = 0;          // wave-uniform
    int first = 0;        // wave-uniform (first qualifying index)
    for (int j0 = 0; j0 < N && cnt < K; j0 += 32) {
        int j = j0 + lane;
        bool hit = false;
        if (j < N) {
            float dx = xyz[j * 3 + 0] - cx;
            float dy = xyz[j * 3 + 1] - cy;
            float dz = xyz[j * 3 + 2] - cz;
            hit = (dx * dx + dy * dy + dz * dz < r2);
        }
        unsigned mask = (unsigned)__ballot(hit);
        if (cnt == 0 && mask != 0u) first = j0 + (__ffs(mask) - 1);
        int pos = __popc(mask & ((1u << lane) - 1u));
        if (hit) {
            int slot = cnt + pos;
            if (slot < K) row[slot] = j;
        }
        cnt += __popc(mask);
    }
    int total = (cnt < K) ? cnt : K;
    for (int c = total + lane; c < K; c += 32) row[c] = first;
}

// ---------------------------------------------------------------------------
// Grouping (per batch): build GEMM input G[S*K][Kpad] (f16, channel-contiguous)
//   cols 0..2  : xyz[j] - new_xyz[s]
//   cols 3..   : gathered features
//   rest       : zero padding
// ---------------------------------------------------------------------------
__global__ void group_kernel(const float* __restrict__ xyz,
                             const float* __restrict__ nxyz,
                             const _Float16* __restrict__ feat, // may be null
                             const int* __restrict__ idx,
                             _Float16* __restrict__ G,
                             int S, int K, int Cfeat, int CfeatStride, int Kpad)
{
    int row = blockIdx.x;             // row in [0, S*K)
    int s = row / K;
    int j = idx[row];
    _Float16* g = G + (size_t)row * Kpad;
    for (int col = threadIdx.x; col < Kpad; col += blockDim.x) {
        _Float16 v;
        if (col < 3)
            v = (_Float16)(xyz[(size_t)j * 3 + col] - nxyz[(size_t)s * 3 + col]);
        else if (col - 3 < Cfeat)
            v = feat[(size_t)j * CfeatStride + (col - 3)];
        else
            v = (_Float16)0.0f;
        g[col] = v;
    }
}

// ---------------------------------------------------------------------------
// WMMA GEMM with fused (BN+bias) epilogue:  Y = act(alpha[o]*(W@X) + gamma[o])
//   X : [N][Kpad]   f16, row = sample, channel-contiguous
//   W : [Mpad][Kpad] f16 (padded, pre-packed)
//   Yh: [N][Mpad]   f16 output (optional)
//   Yf: [N][Mpad]   f32 output (optional)
// One wave computes a 16(M) x 64(N) tile (4 accumulators, A reuse).
// Block = 128 threads = 4 waves covering 256 output columns.
// Branch-free inner loop so the compiler can clause loads and use partial
// s_wait_loadcnt to pipeline global_load_b128 with v_wmma.
// ---------------------------------------------------------------------------
__global__ void wmma_gemm_kernel(const _Float16* __restrict__ X,
                                 const _Float16* __restrict__ W,
                                 const float* __restrict__ alpha,
                                 const float* __restrict__ gamma,
                                 _Float16* __restrict__ Yh,
                                 float* __restrict__ Yf,
                                 int N, int Kpad, int Mpad, int relu)
{
    const int lane  = threadIdx.x & 31;
    const int wave  = threadIdx.x >> 5;
    const int m0    = blockIdx.y * 16;
    const long n0   = ((long)blockIdx.x * 4 + wave) * 64;
    if (n0 >= N) return;                       // wave-uniform: EXEC stays full

    const int  lh = lane & 15;
    const bool hi = (lane >= 16);

    v8f acc[4] = {};

    const int m = m0 + lh;                     // A-matrix row for this lane
    for (int k0 = 0; k0 < Kpad; k0 += 32) {
        // ---- A fragment: 16-bit A 16x32 layout (ISA 7.12.2) ----
        // lanes 0-15: halves 0..7 = K 0..7,  halves 8..15 = K 16..23
        // lanes16-31: halves 0..7 = K 8..15, halves 8..15 = K 24..31
        const _Float16* wr = W + (size_t)m * Kpad + k0 + (hi ? 8 : 0);
        h8f alo = *(const h8f*)(wr);
        h8f ahi = *(const h8f*)(wr + 16);
        v16h a;
        #pragma unroll
        for (int i = 0; i < 8; ++i) { a[i] = alo[i]; a[i + 8] = ahi[i]; }

        #pragma unroll
        for (int t = 0; t < 4; ++t) {
            // ---- B fragment: 16-bit B 32x16 ----
            // lanes 0-15: col n = lane,    K = 0..7, 16..23
            // lanes16-31: col n = lane-16, K = 8..15, 24..31
            long n = n0 + t * 16 + lh;
            if (n >= N) n = N - 1;             // clamp (tiles always full here)
            const _Float16* xr = X + (size_t)n * Kpad + k0 + (hi ? 8 : 0);
            h8f blo = *(const h8f*)(xr);
            h8f bhi = *(const h8f*)(xr + 16);
            v16h bb;
            #pragma unroll
            for (int i = 0; i < 8; ++i) { bb[i] = blo[i]; bb[i + 8] = bhi[i]; }

            acc[t] = __builtin_amdgcn_wmma_f32_16x16x32_f16(
                false, a, false, bb, (short)0, acc[t], false, false);
        }
    }

    // ---- epilogue: D layout -> lane holds col n=lh, rows mb..mb+7 ----
    const int mb = m0 + (hi ? 8 : 0);
    float al[8], gm[8];
    #pragma unroll
    for (int r = 0; r < 8; ++r) { al[r] = alpha[mb + r]; gm[r] = gamma[mb + r]; }

    #pragma unroll
    for (int t = 0; t < 4; ++t) {
        long n = n0 + t * 16 + lh;
        if (n >= N) continue;
        float v[8];
        #pragma unroll
        for (int r = 0; r < 8; ++r) {
            float y = acc[t][r] * al[r] + gm[r];
            if (relu) y = fmaxf(y, 0.0f);
            v[r] = y;
        }
        if (Yh) {
            h8f o;
            #pragma unroll
            for (int r = 0; r < 8; ++r) o[r] = (_Float16)v[r];
            *(h8f*)(Yh + (size_t)n * Mpad + mb) = o;   // 16B contiguous store
        }
        if (Yf) {
            #pragma unroll
            for (int r = 0; r < 8; ++r) Yf[(size_t)n * Mpad + mb + r] = v[r];
        }
    }
}

// ---------------------------------------------------------------------------
// Max over K samples (per batch): feat[s][cOff+c] = max_k Y[(s*K+k)][c]
// ---------------------------------------------------------------------------
__global__ void maxpool_kernel(const _Float16* __restrict__ Y,
                               _Float16* __restrict__ feat,
                               int S, int K, int Mpad, int Cout, int cOff, int Cstride)
{
    int s = blockIdx.x;
    for (int c = threadIdx.x; c < Cout; c += blockDim.x) {
        const _Float16* y = Y + (size_t)s * K * Mpad + c;
        float m = -1e30f;
        for (int k = 0; k < K; ++k) m = fmaxf(m, (float)y[(size_t)k * Mpad]);
        feat[(size_t)s * Cstride + cOff + c] = (_Float16)m;
    }
}

// ---------------------------------------------------------------------------
// Pack f32 weights into padded f16 [Mpad][Kpad]
// ---------------------------------------------------------------------------
__global__ void pack_w_kernel(const float* __restrict__ W, _Float16* __restrict__ Wp,
                              int Cout, int Cin, int Mpad, int Kpad)
{
    int i = blockIdx.x * blockDim.x + threadIdx.x;
    if (i >= Mpad * Kpad) return;
    int m = i / Kpad, k = i % Kpad;
    float v = (m < Cout && k < Cin) ? W[(size_t)m * Cin + k] : 0.0f;
    Wp[i] = (_Float16)v;
}

// ---------------------------------------------------------------------------
// Fold bias+BN into per-channel alpha/gamma:
//   y = ((Wx + b) - mu)*g*rsqrt(var+eps) + beta = alpha*(Wx) + gamma
// ---------------------------------------------------------------------------
__global__ void bn_coef_kernel(const float* __restrict__ b, const float* __restrict__ g,
                               const float* __restrict__ be, const float* __restrict__ mu,
                               const float* __restrict__ var,
                               float* __restrict__ alpha, float* __restrict__ gamma,
                               int Cout, int Mpad)
{
    int o = blockIdx.x * blockDim.x + threadIdx.x;
    if (o >= Mpad) return;
    if (o < Cout) {
        float a = g[o] * rsqrtf(var[o] + BN_EPS);
        alpha[o] = a;
        gamma[o] = (b[o] - mu[o]) * a + be[o];
    } else { alpha[o] = 0.0f; gamma[o] = 0.0f; }
}

// ---------------------------------------------------------------------------
// Softmax over the point axis: logits Yf[b*S+s][o] -> Wsoft[b][o][s]
// One block per (b,o).
// ---------------------------------------------------------------------------
__global__ void softmax_kernel(const float* __restrict__ Yf, float* __restrict__ Wsoft,
                               int S, int Mpad, int Cout)
{
    int bo = blockIdx.x;
    int b = bo / Cout, o = bo % Cout;
    __shared__ float red[256];
    const float* base = Yf + (size_t)b * S * Mpad + o;
    int t = threadIdx.x;
    float mx = -1e30f;
    for (int s = t; s < S; s += 256) mx = fmaxf(mx, base[(size_t)s * Mpad]);
    red[t] = mx; __syncthreads();
    for (int w = 128; w > 0; w >>= 1) {
        if (t < w) red[t] = fmaxf(red[t], red[t + w]);
        __syncthreads();
    }
    mx = red[0]; __syncthreads();
    float sum = 0.0f;
    for (int s = t; s < S; s += 256) sum += expf(base[(size_t)s * Mpad] - mx);
    red[t] = sum; __syncthreads();
    for (int w = 128; w > 0; w >>= 1) {
        if (t < w) red[t] += red[t + w];
        __syncthreads();
    }
    float inv = 1.0f / red[0];
    float* out = Wsoft + (size_t)bo * S;
    for (int s = t; s < S; s += 256) out[s] = expf(base[(size_t)s * Mpad] - mx) * inv;
}

// ---------------------------------------------------------------------------
// skel[b][k][d] = sum_s Wsoft[b][k][s] * xyz[b][s][d]
// ---------------------------------------------------------------------------
__global__ void skel_kernel(const float* __restrict__ Wsoft, const float* __restrict__ xyz,
                            float* __restrict__ out, int Kc, int S, int total)
{
    int i = blockIdx.x * blockDim.x + threadIdx.x;
    if (i >= total) return;
    int d = i % 3, bk = i / 3, b = bk / Kc;
    const float* w = Wsoft + (size_t)bk * S;
    const float* x = xyz + (size_t)b * S * 3 + d;
    float acc = 0.0f;
    for (int s = 0; s < S; ++s) acc += w[s] * x[(size_t)s * 3];
    out[i] = acc;
}

// ---------------------------------------------------------------------------
// shape[b][k][c] = sum_s feat[b][s][c] * Wsoft[b][k][s]
// One block per (b,k); w row staged in LDS; coalesced feat reads.
// ---------------------------------------------------------------------------
__global__ void shape_kernel(const _Float16* __restrict__ feat, const float* __restrict__ Wsoft,
                             float* __restrict__ out, int Kc, int S, int C)
{
    int bk = blockIdx.x;
    int b = bk / Kc;
    __shared__ float w[512];
    for (int s = threadIdx.x; s < S; s += blockDim.x) w[s] = Wsoft[(size_t)bk * S + s];
    __syncthreads();
    const _Float16* f = feat + (size_t)b * S * C;
    for (int c = threadIdx.x; c < C; c += blockDim.x) {
        float acc = 0.0f;
        for (int s = 0; s < S; ++s) acc += (float)f[(size_t)s * C + c] * w[s];
        out[(size_t)bk * C + c] = acc;
    }
}

// ---------------------------------------------------------------------------
// Host orchestration
// ---------------------------------------------------------------------------
extern "C" void kernel_launch(void* const* d_in, const int* in_sizes, int n_in,
                              void* d_out, int out_size, void* d_ws, size_t ws_size,
                              hipStream_t stream)
{
    (void)in_sizes; (void)n_in; (void)out_size; (void)ws_size;

    static const int   sa_np[4]      = {1024, 768, 512, 512};
    static const float sa_r[4][2]    = {{0.1f, 0.2f}, {0.2f, 0.4f}, {0.4f, 0.6f}, {0.6f, 0.8f}};
    static const int   sa_k[4][2]    = {{16, 32}, {32, 64}, {32, 64}, {64, 128}};
    static const int   sa_dims[4][4] = {{3, 16, 16, 32}, {67, 32, 32, 64},
                                        {131, 64, 64, 128}, {259, 128, 128, 256}};
    static const int   head_dims[6]  = {512, 384, 256, 256, 128, 100};
    const int B = 16, N0 = 2048, HS = 512, KC = 100;

    // -------- workspace carving (deterministic bump allocator) --------
    char* ws = (char*)d_ws;
    size_t off = 0;
    auto alloc = [&](size_t bytes) -> void* {
        off = (off + 255) & ~(size_t)255;
        void* p = ws + off;
        off += bytes;
        return p;
    };

    float*     xyzbuf[2]; xyzbuf[0] = (float*)alloc((size_t)B * 1024 * 3 * 4);
                          xyzbuf[1] = (float*)alloc((size_t)B * 1024 * 3 * 4);
    int*       fidx    = (int*)alloc((size_t)B * 1024 * 4);
    int*       ballidx = (int*)alloc((size_t)65536 * 4);
    _Float16*  featbuf[2]; featbuf[0] = (_Float16*)alloc((size_t)B * 512 * 512 * 2);
                           featbuf[1] = (_Float16*)alloc((size_t)B * 512 * 512 * 2);
    _Float16*  G       = (_Float16*)alloc((size_t)65536 * 288 * 2);
    _Float16*  actbuf[2]; actbuf[0] = (_Float16*)alloc((size_t)65536 * 256 * 2);
                          actbuf[1] = (_Float16*)alloc((size_t)65536 * 256 * 2);
    _Float16*  headbuf[2]; headbuf[0] = (_Float16*)alloc((size_t)B * HS * 512 * 2);
                           headbuf[1] = (_Float16*)alloc((size_t)B * HS * 512 * 2);
    float*     Yf32    = (float*)alloc((size_t)B * HS * 128 * 4);
    float*     Wsoft   = (float*)alloc((size_t)B * KC * HS * 4);

    // -------- pack all weights + fold BN coefficients --------
    _Float16* WpSA[4][2][3]; float* alSA[4][2][3]; float* gmSA[4][2][3];
    _Float16* WpH[5];        float* alH[5];        float* gmH[5];

    for (int st = 0; st < 4; ++st)
        for (int sc = 0; sc < 2; ++sc)
            for (int l = 0; l < 3; ++l) {
                int cin = sa_dims[st][l], cout = sa_dims[st][l + 1];
                int Kp = round32(cin), Mp = round32(cout);
                WpSA[st][sc][l] = (_Float16*)alloc((size_t)Mp * Kp * 2);
                alSA[st][sc][l] = (float*)alloc((size_t)Mp * 4);
                gmSA[st][sc][l] = (float*)alloc((size_t)Mp * 4);
                int base = 1 + (((st * 2 + sc) * 3) + l) * 6;
                int tot = Mp * Kp;
                pack_w_kernel<<<(tot + 255) / 256, 256, 0, stream>>>(
                    (const float*)d_in[base + 0], WpSA[st][sc][l], cout, cin, Mp, Kp);
                bn_coef_kernel<<<(Mp + 63) / 64, 64, 0, stream>>>(
                    (const float*)d_in[base + 1], (const float*)d_in[base + 2],
                    (const float*)d_in[base + 3], (const float*)d_in[base + 4],
                    (const float*)d_in[base + 5], alSA[st][sc][l], gmSA[st][sc][l], cout, Mp);
            }
    for (int l = 0; l < 5; ++l) {
        int cin = head_dims[l], cout = head_dims[l + 1];
        int Kp = round32(cin), Mp = round32(cout);
        WpH[l] = (_Float16*)alloc((size_t)Mp * Kp * 2);
        alH[l] = (float*)alloc((size_t)Mp * 4);
        gmH[l] = (float*)alloc((size_t)Mp * 4);
        int base = 1 + 144 + l * 6;
        int tot = Mp * Kp;
        pack_w_kernel<<<(tot + 255) / 256, 256, 0, stream>>>(
            (const float*)d_in[base + 0], WpH[l], cout, cin, Mp, Kp);
        bn_coef_kernel<<<(Mp + 63) / 64, 64, 0, stream>>>(
            (const float*)d_in[base + 1], (const float*)d_in[base + 2],
            (const float*)d_in[base + 3], (const float*)d_in[base + 4],
            (const float*)d_in[base + 5], alH[l], gmH[l], cout, Mp);
    }

    auto launch_gemm = [&](const _Float16* X, const _Float16* Wp, const float* al,
                           const float* gm, _Float16* Yh, float* Yf,
                           long N, int Kpad, int Mpad, int relu) {
        dim3 grid((unsigned)((N + 255) / 256), (unsigned)(Mpad / 16), 1);
        wmma_gemm_kernel<<<grid, 128, 0, stream>>>(X, Wp, al, gm, Yh, Yf,
                                                   (int)N, Kpad, Mpad, relu);
    };

    // -------- SA stages --------
    const float* xyz_cur = (const float*)d_in[0];   // [B][2048][3]
    int N_cur = N0;
    const _Float16* feat_cur = nullptr;
    int C_cur = 0;
    int xb = 0, fb = 0;

    for (int st = 0; st < 4; ++st) {
        int np = sa_np[st];
        fps_kernel<<<B, FPS_THREADS, 0, stream>>>(xyz_cur, N_cur, np, fidx);
        float* nxyz = xyzbuf[xb];
        gather_xyz_kernel<<<dim3((np + 255) / 256, B), 256, 0, stream>>>(
            xyz_cur, fidx, nxyz, N_cur, np);

        _Float16* feat_out = featbuf[fb];
        int CoutL = sa_dims[st][3];
        int Ctot  = 2 * CoutL;
        int Cin   = sa_dims[st][0];
        int Kpad1 = round32(Cin);

        for (int sc = 0; sc < 2; ++sc) {
            int   Ks = sa_k[st][sc];
            float r  = sa_r[st][sc];
            for (int b = 0; b < B; ++b) {
                const float* xb_ptr = xyz_cur + (size_t)b * N_cur * 3;
                const float* nx_ptr = nxyz + (size_t)b * np * 3;
                long Nrows = (long)np * Ks;

                ball_query_kernel<<<np, 32, 0, stream>>>(
                    xb_ptr, nx_ptr, N_cur, np, r * r, Ks, ballidx);

                const _Float16* fptr =
                    feat_cur ? feat_cur + (size_t)b * N_cur * C_cur : nullptr;
                group_kernel<<<(unsigned)Nrows, 64, 0, stream>>>(
                    xb_ptr, nx_ptr, fptr, ballidx, G, np, Ks, C_cur, C_cur, Kpad1);

                const _Float16* Xl = G;
                int Kp = Kpad1;
                _Float16* Yl = nullptr;
                for (int l = 0; l < 3; ++l) {
                    int Mp = round32(sa_dims[st][l + 1]);
                    Yl = actbuf[l & 1];
                    launch_gemm(Xl, WpSA[st][sc][l], alSA[st][sc][l], gmSA[st][sc][l],
                                Yl, nullptr, Nrows, Kp, Mp, 1);
                    Xl = Yl; Kp = Mp;
                }
                int MpLast = round32(CoutL);
                maxpool_kernel<<<np, 128, 0, stream>>>(
                    Xl, feat_out + (size_t)b * np * Ctot,
                    np, Ks, MpLast, CoutL, sc * CoutL, Ctot);
            }
        }
        xyz_cur = nxyz; N_cur = np;
        feat_cur = feat_out; C_cur = Ctot;
        xb ^= 1; fb ^= 1;
    }

    // -------- head MLP (batch-fused: N = B*512 rows) --------
    const _Float16* hx = feat_cur;      // [B*512][512]
    long Nh = (long)B * HS;
    int Kp = C_cur;                     // 512
    for (int l = 0; l < 5; ++l) {
        int Mp = round32(head_dims[l + 1]);
        bool last = (l == 4);
        _Float16* Yl = headbuf[l & 1];
        launch_gemm(hx, WpH[l], alH[l], gmH[l],
                    last ? nullptr : Yl, last ? Yf32 : nullptr,
                    Nh, Kp, Mp, last ? 0 : 1);
        hx = Yl; Kp = Mp;
    }

    // -------- softmax over points, then final einsums --------
    softmax_kernel<<<B * KC, 256, 0, stream>>>(Yf32, Wsoft, HS, 128, KC);

    float* out_skel  = (float*)d_out;                 // [B][100][3]
    float* out_shape = out_skel + (size_t)B * KC * 3; // [B][100][512]
    int total_skel = B * KC * 3;
    skel_kernel<<<(total_skel + 255) / 256, 256, 0, stream>>>(
        Wsoft, xyz_cur, out_skel, KC, HS, total_skel);
    shape_kernel<<<B * KC, 256, 0, stream>>>(
        feat_cur, Wsoft, out_shape, KC, HS, C_cur);
}
